// Model_75256416960730
// MI455X (gfx1250) — matrix operand order repaired and checked by reading, not compiled, not run on hardware
//
#include <hip/hip_runtime.h>
#include <hip/hip_bf16.h>
#include <math.h>

#define D_MODEL  512
#define D_OUT    256
#define N_STATE  64
#define N_LAYERS 4
#define BATCH    16
#define SEQLEN   2048
#define MROWS    (BATCH * SEQLEN)   // 32768 token rows
#define L2       (2 * SEQLEN)       // 4096 FFT length

typedef __attribute__((ext_vector_type(16))) __bf16 v16bf;
typedef __attribute__((ext_vector_type(8)))  float  v8f;

// ---------------------------------------------------------------------------
// helpers
// ---------------------------------------------------------------------------
__device__ __forceinline__ unsigned short f2bf(float f) {
  unsigned int u = __float_as_uint(f);
  unsigned int r = (u + 0x7FFFu + ((u >> 16) & 1u)) >> 16;  // RNE
  return (unsigned short)r;
}

__device__ __forceinline__ float gelu_f(float v) {
  float c = 0.7978845608028654f * (v + 0.044715f * v * v * v);
  return 0.5f * v * (1.0f + tanhf(c));
}

// ---------------------------------------------------------------------------
// bf16 WMMA GEMM:  Y[M,N] = Abf[M,K](bf16) * WT[N,K](bf16)^T + bias (+ skip)
// block = 256 threads = 8 waves (4 M x 2 N); block tile 128 x 128;
// wave tile 32(M) x 64(N) -> 2x4 v_wmma_f32_16x16x32_bf16 per K-step.
// All fragments are straight 16B loads of pre-packed bf16 (no in-loop cvt).
// ---------------------------------------------------------------------------
__global__ __launch_bounds__(256)
void gemm_wmma_bf16(const unsigned short* __restrict__ Abf,
                    const unsigned short* __restrict__ WT,
                    const float* __restrict__ bias,
                    const float* __restrict__ skip,
                    float* __restrict__ Y,
                    int Mdim, int Ndim, int Kdim) {
  const int lane = threadIdx.x & 31;
  const int wave = threadIdx.x >> 5;
  const int wm = wave & 3;                 // 0..3 -> M
  const int wn = wave >> 2;                // 0..1 -> N
  const int m0 = blockIdx.y * 128 + wm * 32;
  const int n0 = blockIdx.x * 128 + wn * 64;
  const int lh = lane >> 4;                // lane half 0/1
  const int ll = lane & 15;

  v8f acc[2][4];
  const v8f vzero = {0.f, 0.f, 0.f, 0.f, 0.f, 0.f, 0.f, 0.f};
#pragma unroll
  for (int i = 0; i < 2; ++i)
#pragma unroll
    for (int j = 0; j < 4; ++j) acc[i][j] = vzero;

  // per-lane base pointers (A rows / B rows fixed per lane)
  const unsigned short* aBase0 = Abf + (size_t)(m0 + ll) * Kdim;
  const unsigned short* aBase1 = Abf + (size_t)(m0 + 16 + ll) * Kdim;

  for (int kb = 0; kb < Kdim; kb += 32) {
    // ---- A fragments: 16x32 bf16 layout (ISA 7.12.2) ----
    // lane needs K = kb + 8*lanehalf + [0..7]  and  +16
    union { v16bf v; uint4 q[2]; } afrag[2];
    {
      const int koff = kb + (lh << 3);
      afrag[0].q[0] = *(const uint4*)(aBase0 + koff);
      afrag[0].q[1] = *(const uint4*)(aBase0 + koff + 16);
      afrag[1].q[0] = *(const uint4*)(aBase1 + koff);
      afrag[1].q[1] = *(const uint4*)(aBase1 + koff + 16);
    }
    // ---- B fragments: W^T rows contiguous in K -> two 16B loads each ----
    union { v16bf v; uint4 q[2]; } bfrag[4];
#pragma unroll
    for (int ni = 0; ni < 4; ++ni) {
      const int col = n0 + ni * 16 + ll;
      const unsigned short* bp = WT + (size_t)col * Kdim + kb + (lh << 3);
      bfrag[ni].q[0] = *(const uint4*)(bp);
      bfrag[ni].q[1] = *(const uint4*)(bp + 16);
    }
#pragma unroll
    for (int mi = 0; mi < 2; ++mi)
#pragma unroll
      for (int ni = 0; ni < 4; ++ni)
        acc[mi][ni] = __builtin_amdgcn_wmma_f32_16x16x32_bf16(
            false, afrag[mi].v, false, bfrag[ni].v,
            (short)0, acc[mi][ni], false, false);
  }

  // ---- epilogue: C/D layout -> row = v + 8*lanehalf, col = lane%16 ----
#pragma unroll
  for (int mi = 0; mi < 2; ++mi)
#pragma unroll
    for (int ni = 0; ni < 4; ++ni) {
      const int col = n0 + ni * 16 + ll;
      const float bv = bias ? bias[col] : 0.0f;
#pragma unroll
      for (int v = 0; v < 8; ++v) {
        const int row = m0 + mi * 16 + v + (lh << 3);
        float val = acc[mi][ni][v] + bv;
        if (skip) val += skip[(size_t)row * Ndim + col];
        Y[(size_t)row * Ndim + col] = val;
      }
    }
}

// ---------------------------------------------------------------------------
// weight conversion: W[K,N] f32 -> WT[N,K] bf16
// ---------------------------------------------------------------------------
__global__ void convert_wT_kernel(const float* __restrict__ W,
                                  unsigned short* __restrict__ WT,
                                  int K, int N) {
  int idx = blockIdx.x * 256 + threadIdx.x;
  if (idx < K * N) {
    int k = idx / N, n = idx % N;
    WT[(size_t)n * K + k] = f2bf(W[idx]);
  }
}

// ---------------------------------------------------------------------------
// activation conversion: f32 -> bf16 (8 elements/thread, vectorized)
// ---------------------------------------------------------------------------
__global__ void convert_bf16_kernel(const float* __restrict__ src,
                                    unsigned short* __restrict__ dst,
                                    int n8) {   // n8 = n / 8
  int i = blockIdx.x * 256 + threadIdx.x;
  if (i >= n8) return;
  const float4 a = ((const float4*)src)[2 * i + 0];
  const float4 b = ((const float4*)src)[2 * i + 1];
  union { uint4 q; unsigned short s[8]; } o;
  o.s[0] = f2bf(a.x); o.s[1] = f2bf(a.y); o.s[2] = f2bf(a.z); o.s[3] = f2bf(a.w);
  o.s[4] = f2bf(b.x); o.s[5] = f2bf(b.y); o.s[6] = f2bf(b.z); o.s[7] = f2bf(b.w);
  ((uint4*)dst)[i] = o.q;
}

// ---------------------------------------------------------------------------
// fused GELU + f32->bf16 conversion (GEMM consumes only the bf16 result)
// ---------------------------------------------------------------------------
__global__ void gelu_bf16_kernel(const float* __restrict__ src,
                                 unsigned short* __restrict__ dst,
                                 int n8) {
  int i = blockIdx.x * 256 + threadIdx.x;
  if (i >= n8) return;
  const float4 a = ((const float4*)src)[2 * i + 0];
  const float4 b = ((const float4*)src)[2 * i + 1];
  union { uint4 q; unsigned short s[8]; } o;
  o.s[0] = f2bf(gelu_f(a.x)); o.s[1] = f2bf(gelu_f(a.y));
  o.s[2] = f2bf(gelu_f(a.z)); o.s[3] = f2bf(gelu_f(a.w));
  o.s[4] = f2bf(gelu_f(b.x)); o.s[5] = f2bf(gelu_f(b.y));
  o.s[6] = f2bf(gelu_f(b.z)); o.s[7] = f2bf(gelu_f(b.w));
  ((uint4*)dst)[i] = o.q;
}

// ---------------------------------------------------------------------------
// LayerNorm over D_MODEL=512, one block(256) per row
// ---------------------------------------------------------------------------
__global__ __launch_bounds__(256)
void layernorm_kernel(const float* __restrict__ x,
                      const float* __restrict__ scale,
                      const float* __restrict__ bias,
                      float* __restrict__ y) {
  const int row = blockIdx.x;
  const int tid = threadIdx.x;
  const float* xr = x + (size_t)row * D_MODEL;
  float v0 = xr[tid], v1 = xr[tid + 256];
  __shared__ float red[256];
  red[tid] = v0 + v1; __syncthreads();
  for (int s = 128; s > 0; s >>= 1) { if (tid < s) red[tid] += red[tid + s]; __syncthreads(); }
  const float mu = red[0] * (1.0f / 512.0f);
  __syncthreads();
  const float d0 = v0 - mu, d1 = v1 - mu;
  red[tid] = d0 * d0 + d1 * d1; __syncthreads();
  for (int s = 128; s > 0; s >>= 1) { if (tid < s) red[tid] += red[tid + s]; __syncthreads(); }
  const float rstd = rsqrtf(red[0] * (1.0f / 512.0f) + 1e-6f);
  float* yr = y + (size_t)row * D_MODEL;
  yr[tid]       = d0 * rstd * scale[tid]       + bias[tid];
  yr[tid + 256] = d1 * rstd * scale[tid + 256] + bias[tid + 256];
}

// ---------------------------------------------------------------------------
// S4 per-layer coefficient precompute: conj(C)B, conj(C)P, conj(P)B, |P|^2, Lam
// ---------------------------------------------------------------------------
__global__ void s4_coeff_kernel(const float* lre, const float* lim,
                                const float* pre, const float* pim,
                                const float* bre, const float* bim,
                                const float* cre, const float* cim,
                                float* __restrict__ coeff /*[H][N][10]*/) {
  int idx = blockIdx.x * 256 + threadIdx.x;
  if (idx >= D_MODEL * N_STATE) return;
  float br = bre[idx], bi = bim[idx];
  float cr = cre[idx], ci = cim[idx];
  float pr = pre[idx], pi = pim[idx];
  float* o = coeff + (size_t)idx * 10;
  o[0] = cr * br + ci * bi;  o[1] = cr * bi - ci * br;   // conj(C)*B
  o[2] = cr * pr + ci * pi;  o[3] = cr * pi - ci * pr;   // conj(C)*P
  o[4] = pr * br + pi * bi;  o[5] = pr * bi - pi * br;   // conj(P)*B
  o[6] = pr * pr + pi * pi;  o[7] = 0.0f;                // conj(P)*P (real)
  o[8] = lre[idx];           o[9] = lim[idx];
}

// ---------------------------------------------------------------------------
// Cauchy resolvent: at_roots[h][l]  (grid: (L/256, H))
// ---------------------------------------------------------------------------
__global__ __launch_bounds__(256)
void cauchy_kernel(const float* __restrict__ coeff,
                   const float* __restrict__ log_step,  // [H] layer slice
                   float* __restrict__ ar /*[H][L][2]*/) {
  const int h = blockIdx.y;
  const int l = blockIdx.x * 256 + threadIdx.x;
  __shared__ float cf[N_STATE * 10];
  for (int i = threadIdx.x; i < N_STATE * 10; i += 256)
    cf[i] = coeff[(size_t)h * N_STATE * 10 + i];
  __syncthreads();

  const float step  = __expf(log_step[h]);
  const float theta = (6.283185307179586f / (float)SEQLEN) * (float)l;
  float sn, cs; __sincosf(theta, &sn, &cs);
  // Omega = exp(-i theta) = (cs, -sn)
  const float opr = 1.0f + cs, opi = -sn;     // 1 + Omega
  const float omr = 1.0f - cs, omi =  sn;     // 1 - Omega
  const float inv = 1.0f / (opr * opr + opi * opi);
  const float ipr = opr * inv, ipi = -opi * inv;  // 1/(1+Omega)
  const float c_r = 2.0f * ipr, c_i = 2.0f * ipi;
  const float gs  = 2.0f / step;
  const float gr  = gs * (omr * ipr - omi * ipi);
  const float gi  = gs * (omr * ipi + omi * ipr);

  float k00r = 0.f, k00i = 0.f, k01r = 0.f, k01i = 0.f;
  float k10r = 0.f, k10i = 0.f, k11r = 0.f, k11i = 0.f;
#pragma unroll 4
  for (int n = 0; n < N_STATE; ++n) {
    const float* o = cf + n * 10;
    const float dr = gr - o[8], di = gi - o[9];
    const float rn = 1.0f / (dr * dr + di * di);
    const float rr = dr * rn, ri = -di * rn;         // 1/(g - Lam)
    k00r += o[0] * rr - o[1] * ri;  k00i += o[0] * ri + o[1] * rr;
    k01r += o[2] * rr - o[3] * ri;  k01i += o[2] * ri + o[3] * rr;
    k10r += o[4] * rr - o[5] * ri;  k10i += o[4] * ri + o[5] * rr;
    k11r += o[6] * rr;              k11i += o[6] * ri;
  }
  const float er = 1.0f + k11r, ei = k11i;
  const float en = 1.0f / (er * er + ei * ei);
  const float ier = er * en, iei = -ei * en;         // 1/(1+k11)
  const float t1r = k01r * ier - k01i * iei, t1i = k01r * iei + k01i * ier;
  const float tr  = t1r * k10r - t1i * k10i, ti  = t1r * k10i + t1i * k10r;
  const float ur  = k00r - tr,  ui = k00i - ti;
  float2* out = (float2*)ar;
  out[(size_t)h * SEQLEN + l] =
      make_float2(c_r * ur - c_i * ui, c_r * ui + c_i * ur);
}

// ---------------------------------------------------------------------------
// Stockham radix-2 FFT in LDS (ping-pong).  dir = -1 forward, +1 inverse
// (inverse is unscaled; caller multiplies by 1/n).  Returns result buffer.
// ---------------------------------------------------------------------------
__device__ float2* fft_lds(float2* bufA, float2* bufB,
                           int n, int logn, float dir, int tid, int nt) {
  float2* src = bufA;
  float2* dst = bufB;
  const int half = n >> 1;
  for (int st = 0; st < logn; ++st) {
    const int s = 1 << st;
    const float th0 = dir * 6.283185307179586f / (float)(n >> st);
    for (int t = tid; t < half; t += nt) {
      const int q = t & (s - 1);
      const int p = t >> st;
      const float2 a = src[t];
      const float2 b = src[t + half];
      float sn, cs; __sincosf(th0 * (float)p, &sn, &cs);
      const float dxr = a.x - b.x, dxi = a.y - b.y;
      dst[q + ((2 * p) << st)]     = make_float2(a.x + b.x, a.y + b.y);
      dst[q + ((2 * p + 1) << st)] = make_float2(cs * dxr - sn * dxi,
                                                 cs * dxi + sn * dxr);
    }
    __syncthreads();
    float2* tmp = src; src = dst; dst = tmp;
  }
  return src;
}

// ---------------------------------------------------------------------------
// K-hat: per channel h, K = real(ifft_L(at_roots)); Khat = fft_{2L}(pad(K))
// dynamic LDS: 2 * 4096 * sizeof(float2) = 64KB
// ---------------------------------------------------------------------------
__global__ __launch_bounds__(256)
void s4_kernel_fft(const float* __restrict__ ar, float* __restrict__ khat) {
  extern __shared__ float2 sm[];
  float2* A = sm;
  float2* B = sm + L2;
  const int h = blockIdx.x;
  const int tid = threadIdx.x;
  const float2* arh = (const float2*)ar + (size_t)h * SEQLEN;
  for (int t = tid; t < SEQLEN; t += 256) A[t] = arh[t];
  __syncthreads();
  float2* res = fft_lds(A, B, SEQLEN, 11, +1.0f, tid, 256);   // ifft (unscaled)
  float2* other = (res == A) ? B : A;
  for (int t = tid; t < L2; t += 256) {
    float kr = (t < SEQLEN) ? res[t].x * (1.0f / (float)SEQLEN) : 0.0f;
    other[t] = make_float2(kr, 0.0f);                          // zero-padded real K
  }
  __syncthreads();
  float2* res2 = fft_lds(other, res, L2, 12, -1.0f, tid, 256); // forward FFT
  float2* kh = (float2*)khat + (size_t)h * L2;
  for (int t = tid; t < L2; t += 256) kh[t] = res2[t];
}

// ---------------------------------------------------------------------------
// FFT convolution per (batch, channel): y = irfft(fft(u_pad) * Khat) + D*u
// uT/yT layout: [B][H][L] (contiguous sequences).  dynamic LDS = 64KB
// ---------------------------------------------------------------------------
__global__ __launch_bounds__(256)
void s4_conv_fft(const float* __restrict__ uT,
                 const float* __restrict__ khat,
                 const float* __restrict__ dparam,  // [H] layer slice
                 float* __restrict__ yT) {
  extern __shared__ float2 sm[];
  float2* A = sm;
  float2* B = sm + L2;
  const int bh = blockIdx.x;            // [0, BATCH*H)
  const int h  = bh % D_MODEL;
  const int tid = threadIdx.x;
  const float* u = uT + (size_t)bh * SEQLEN;
  for (int t = tid; t < L2; t += 256)
    A[t] = make_float2((t < SEQLEN) ? u[t] : 0.0f, 0.0f);
  __syncthreads();
  float2* res = fft_lds(A, B, L2, 12, -1.0f, tid, 256);        // forward
  const float2* kh = (const float2*)khat + (size_t)h * L2;
  for (int t = tid; t < L2; t += 256) {
    const float2 x = res[t], k = kh[t];
    res[t] = make_float2(x.x * k.x - x.y * k.y, x.x * k.y + x.y * k.x);
  }
  __syncthreads();
  float2* other = (res == A) ? B : A;
  float2* res2 = fft_lds(res, other, L2, 12, +1.0f, tid, 256); // inverse
  const float dv = dparam[h];
  float* y = yT + (size_t)bh * SEQLEN;
  for (int t = tid; t < SEQLEN; t += 256)
    y[t] = res2[t].x * (1.0f / (float)L2) + dv * u[t];
}

// ---------------------------------------------------------------------------
// tiled transpose per batch: in[b][R][C] -> out[b][C][R]
// ---------------------------------------------------------------------------
__global__ __launch_bounds__(256)
void transpose_kernel(const float* __restrict__ in, float* __restrict__ out,
                      int R, int C) {
  __shared__ float tile[32][33];
  const int b = blockIdx.z;
  const float* ib = in  + (size_t)b * R * C;
  float* ob       = out + (size_t)b * R * C;
  const int c0 = blockIdx.x * 32, r0 = blockIdx.y * 32;
  const int tx = threadIdx.x & 31, ty = threadIdx.x >> 5;
  for (int i = ty; i < 32; i += 8)
    tile[i][tx] = ib[(size_t)(r0 + i) * C + c0 + tx];
  __syncthreads();
  for (int i = ty; i < 32; i += 8)
    ob[(size_t)(c0 + i) * R + r0 + tx] = tile[tx][i];
}

// ---------------------------------------------------------------------------
// log_softmax over D_OUT=256, one block(256) per row
// ---------------------------------------------------------------------------
__global__ __launch_bounds__(256)
void logsoftmax_kernel(const float* __restrict__ logits, float* __restrict__ out) {
  const int row = blockIdx.x;
  const int tid = threadIdx.x;
  __shared__ float red[256];
  const float v = logits[(size_t)row * D_OUT + tid];
  red[tid] = v; __syncthreads();
  for (int s = 128; s > 0; s >>= 1) { if (tid < s) red[tid] = fmaxf(red[tid], red[tid + s]); __syncthreads(); }
  const float mx = red[0]; __syncthreads();
  red[tid] = __expf(v - mx); __syncthreads();
  for (int s = 128; s > 0; s >>= 1) { if (tid < s) red[tid] += red[tid + s]; __syncthreads(); }
  const float lse = logf(red[0]) + mx;
  out[(size_t)row * D_OUT + tid] = v - lse;
}

// ---------------------------------------------------------------------------
// launcher
// ---------------------------------------------------------------------------
extern "C" void kernel_launch(void* const* d_in, const int* in_sizes, int n_in,
                              void* d_out, int out_size, void* d_ws, size_t ws_size,
                              hipStream_t stream) {
  const float* x     = (const float*)d_in[0];
  const float* enc_w = (const float*)d_in[1];
  const float* enc_b = (const float*)d_in[2];
  const float* dec_w = (const float*)d_in[3];
  const float* dec_b = (const float*)d_in[4];
  const float* ln_s  = (const float*)d_in[5];
  const float* ln_b  = (const float*)d_in[6];
  const float* lre   = (const float*)d_in[7];
  const float* lim   = (const float*)d_in[8];
  const float* pre   = (const float*)d_in[9];
  const float* pim   = (const float*)d_in[10];
  const float* bre   = (const float*)d_in[11];
  const float* bim   = (const float*)d_in[12];
  const float* cre   = (const float*)d_in[13];
  const float* cim   = (const float*)d_in[14];
  const float* lstep = (const float*)d_in[15];
  const float* dprm  = (const float*)d_in[16];
  const float* out_w = (const float*)d_in[17];
  const float* out_b = (const float*)d_in[18];
  float* outp = (float*)d_out;

  // workspace carve-out (256B aligned)
  char* w = (char*)d_ws;
  auto alloc = [&](size_t bytes) -> void* {
    void* p = (void*)w;
    w += (bytes + 255) & ~(size_t)255;
    return p;
  };
  float* bufH   = (float*)alloc((size_t)MROWS * D_MODEL * sizeof(float));
  float* bufZ   = (float*)alloc((size_t)MROWS * D_MODEL * sizeof(float));
  float* bufT   = (float*)alloc((size_t)MROWS * D_MODEL * sizeof(float));
  float* bufT2  = (float*)alloc((size_t)MROWS * D_MODEL * sizeof(float));
  float* logits = (float*)alloc((size_t)MROWS * D_OUT * sizeof(float));
  float* arbuf  = (float*)alloc((size_t)D_MODEL * SEQLEN * 2 * sizeof(float));
  float* khat   = (float*)alloc((size_t)D_MODEL * L2 * 2 * sizeof(float));
  float* coeff  = (float*)alloc((size_t)D_MODEL * N_STATE * 10 * sizeof(float));
  unsigned short* bfAct = (unsigned short*)alloc((size_t)MROWS * D_MODEL * 2);
  unsigned short* wTenc = (unsigned short*)alloc((size_t)D_MODEL * D_MODEL * 2);
  unsigned short* wTout = (unsigned short*)alloc((size_t)N_LAYERS * D_MODEL * D_MODEL * 2);
  unsigned short* wTdec = (unsigned short*)alloc((size_t)D_MODEL * D_OUT * 2);

  const int actN8 = (MROWS * D_MODEL) / 8;

  // ---- one-time weight conversion (f32 -> transposed bf16) ----
  {
    const int n = D_MODEL * D_MODEL;
    convert_wT_kernel<<<(n + 255) / 256, 256, 0, stream>>>(enc_w, wTenc, D_MODEL, D_MODEL);
    for (int i = 0; i < N_LAYERS; ++i)
      convert_wT_kernel<<<(n + 255) / 256, 256, 0, stream>>>(
          out_w + (size_t)i * n, wTout + (size_t)i * n, D_MODEL, D_MODEL);
    const int nd = D_MODEL * D_OUT;
    convert_wT_kernel<<<(nd + 255) / 256, 256, 0, stream>>>(dec_w, wTdec, D_MODEL, D_OUT);
  }

  // ---- encoder: bf16(x) @ enc_w + enc_b ----
  convert_bf16_kernel<<<(actN8 + 255) / 256, 256, 0, stream>>>(x, bfAct, actN8);
  gemm_wmma_bf16<<<dim3(D_MODEL / 128, MROWS / 128), 256, 0, stream>>>(
      bfAct, wTenc, enc_b, nullptr, bufH, MROWS, D_MODEL, D_MODEL);

  // ---- S4 layers ----
  const size_t smem_fft = (size_t)2 * L2 * sizeof(float2);   // 64 KB
  for (int i = 0; i < N_LAYERS; ++i) {
    layernorm_kernel<<<MROWS, 256, 0, stream>>>(
        bufH, ln_s + i * D_MODEL, ln_b + i * D_MODEL, bufZ);

    const size_t hn = (size_t)i * D_MODEL * N_STATE;
    s4_coeff_kernel<<<(D_MODEL * N_STATE + 255) / 256, 256, 0, stream>>>(
        lre + hn, lim + hn, pre + hn, pim + hn,
        bre + hn, bim + hn, cre + hn, cim + hn, coeff);

    cauchy_kernel<<<dim3(SEQLEN / 256, D_MODEL), 256, 0, stream>>>(
        coeff, lstep + i * D_MODEL, arbuf);

    s4_kernel_fft<<<D_MODEL, 256, smem_fft, stream>>>(arbuf, khat);

    transpose_kernel<<<dim3(D_MODEL / 32, SEQLEN / 32, BATCH), 256, 0, stream>>>(
        bufZ, bufT, SEQLEN, D_MODEL);

    s4_conv_fft<<<BATCH * D_MODEL, 256, smem_fft, stream>>>(
        bufT, khat, dprm + i * D_MODEL, bufT2);

    transpose_kernel<<<dim3(SEQLEN / 32, D_MODEL / 32, BATCH), 256, 0, stream>>>(
        bufT2, bufZ, D_MODEL, SEQLEN);

    // fused GELU + bf16 pack (GEMM A-operand)
    gelu_bf16_kernel<<<(actN8 + 255) / 256, 256, 0, stream>>>(bufZ, bfAct, actN8);

    // h = skip + gelu(z) @ out_w + out_b   (skip fused into GEMM epilogue)
    gemm_wmma_bf16<<<dim3(D_MODEL / 128, MROWS / 128), 256, 0, stream>>>(
        bfAct, wTout + (size_t)i * D_MODEL * D_MODEL,
        out_b + i * D_MODEL, bufH, bufH, MROWS, D_MODEL, D_MODEL);
  }

  // ---- decoder + log_softmax ----
  convert_bf16_kernel<<<(actN8 + 255) / 256, 256, 0, stream>>>(bufH, bfAct, actN8);
  gemm_wmma_bf16<<<dim3(D_OUT / 128, MROWS / 128), 256, 0, stream>>>(
      bfAct, wTdec, dec_b, nullptr, logits, MROWS, D_OUT, D_MODEL);
  logsoftmax_kernel<<<MROWS, 256, 0, stream>>>(logits, outp);
}
